// MCSPRLoss_2044404433176
// MI455X (gfx1250) — compile-verified, matrix operand order checked by reading, not run on hardware
//
#include <hip/hip_runtime.h>
#include <hip/hip_bf16.h>

typedef __attribute__((ext_vector_type(16))) __bf16        v16bf;
typedef __attribute__((ext_vector_type(8)))  __bf16        v8bf;
typedef __attribute__((ext_vector_type(16))) float         v16f;
typedef __attribute__((ext_vector_type(8)))  float         v8f;
typedef __attribute__((ext_vector_type(4)))  float         v4f;
typedef __attribute__((ext_vector_type(16))) unsigned short v16us;
typedef __attribute__((ext_vector_type(8)))  unsigned short v8us;

#define N_TOT    262144
#define D_IN     128
#define NMOD     64
#define NCTX     16
#define CHUNK    256
#define NCHUNK   (N_TOT / CHUNK)          // 1024
#define NBLOCKS  256
#define ZT_STRIDE 264                     // CHUNK + 8 pad: rows stay 16B aligned, spreads banks

#define S_ELEMS  (NCTX * NMOD * NMOD)     // 65536
#define MU_ELEMS (NCTX * NMOD)            // 1024
#define WS_ELEMS (S_ELEMS + MU_ELEMS + NCTX)

__device__ __forceinline__ v8f vzero8() {
  v8f z = {0.f, 0.f, 0.f, 0.f, 0.f, 0.f, 0.f, 0.f};
  return z;
}
// Load one 16x32 bf16 fragment (A or B side; identical per-lane pattern:
// lanes 0-15 take K = n0..n0+7 and n0+16..n0+23, lanes 16-31 shift K by +8).
__device__ __forceinline__ v16bf ld_frag16(const unsigned short* rowp, int n0, int hi8) {
  v8us lo = *(const v8us*)(rowp + n0 + hi8);
  v8us hv = *(const v8us*)(rowp + n0 + 16 + hi8);
  return __builtin_bit_cast(
      v16bf, __builtin_shufflevector(lo, hv, 0, 1, 2, 3, 4, 5, 6, 7,
                                             8, 9, 10, 11, 12, 13, 14, 15));
}

__global__ void zero_ws_kernel(float* ws) {
  int i = blockIdx.x * blockDim.x + threadIdx.x;
  if (i < WS_ELEMS) ws[i] = 0.0f;
}

__global__ __launch_bounds__(512, 1) void mcspr_accum_kernel(
    const float* __restrict__ Y, const float* __restrict__ CW,
    const float* __restrict__ Mp, float* __restrict__ S,
    float* __restrict__ MU, float* __restrict__ WSUM) {
  __shared__ __align__(16) unsigned short sMp[NMOD * D_IN];        // 16 KB bf16 M_pinv
  __shared__ __align__(16) unsigned short sZT[NMOD][ZT_STRIDE];    // 33 KB bf16 Z^T
  __shared__ __align__(16) unsigned short sWb[NCTX][CHUNK];        // 8 KB  bf16 weights^T
  __shared__ float sMu[NCTX * NMOD];                               // 4 KB  mu partials
  __shared__ float sWsum[NCTX];

  const int tid = threadIdx.x;
  const int wave = tid >> 5;          // 0..15  (16 waves: one n-tile, one context each)
  const int lane = tid & 31;
  const int laneM = lane & 15;
  const int hi8 = ((lane >> 4) & 1) * 8;

  for (int i = tid; i < NMOD * D_IN; i += 512) {
    __bf16 b = (__bf16)Mp[i];                  // hw v_cvt
    sMp[i] = __builtin_bit_cast(unsigned short, b);
  }
  for (int p = tid; p < NCTX * NMOD; p += 512) sMu[p] = 0.0f;
  if (tid < NCTX) sWsum[tid] = 0.0f;
  __syncthreads();

  v8f acc[4][4];                       // 16 S-tiles for context t = wave (f32 accum)
#pragma unroll
  for (int a = 0; a < 4; ++a)
#pragma unroll
    for (int b = 0; b < 4; ++b) acc[a][b] = vzero8();

  for (int c = blockIdx.x; c < NCHUNK; c += gridDim.x) {
    const int nbase = c * CHUNK;
    __syncthreads();                   // previous chunk's readers done with sZT/sWb

    // ---- Phase A: stage weights (bf16) + exact f32 wsum partial per context
    {
      const int t = tid & 15;          // each thread touches a single context
      float wpart = 0.0f;
#pragma unroll
      for (int k = 0; k < 8; ++k) {
        int i = tid + k * 512;         // 0..4095 over (n_local, t)
        int nl = i >> 4;
        float w = CW[(size_t)(nbase + nl) * NCTX + t];
        __bf16 wb = (__bf16)w;
        sWb[t][nl] = __builtin_bit_cast(unsigned short, wb);
        wpart += w;
      }
      atomicAdd(&sWsum[t], wpart);
    }

    // ---- Phase B: Z tile = Y_rows x M_pinv^T via WMMA; store Z^T bf16 to LDS
    {
      const int nt = wave;
      const int row = nbase + nt * 16 + laneM;
      v8f zacc[4];
#pragma unroll
      for (int mt = 0; mt < 4; ++mt) zacc[mt] = vzero8();
#pragma unroll
      for (int ks = 0; ks < 4; ++ks) {
        const int d0 = ks * 32;
        const float* yb = Y + (size_t)row * D_IN + d0 + hi8;
        v4f q0 = *(const v4f*)(yb + 0);
        v4f q1 = *(const v4f*)(yb + 4);
        v4f q2 = *(const v4f*)(yb + 16);
        v4f q3 = *(const v4f*)(yb + 20);
        // batch all four M_pinv fragments first: loads clause together,
        // one dscnt wait, then the 4 WMMAs stream back-to-back
        v16bf bf[4];
#pragma unroll
        for (int mt = 0; mt < 4; ++mt)
          bf[mt] = ld_frag16(&sMp[(mt * 16 + laneM) * D_IN], d0, hi8);
        v16f yf;
#pragma unroll
        for (int j = 0; j < 4; ++j) {
          yf[j] = q0[j];
          yf[4 + j] = q1[j];
          yf[8 + j] = q2[j];
          yf[12 + j] = q3[j];
        }
        v16bf af = __builtin_convertvector(yf, v16bf);   // packed v_cvt f32->bf16
#pragma unroll
        for (int mt = 0; mt < 4; ++mt)
          zacc[mt] = __builtin_amdgcn_wmma_f32_16x16x32_bf16(
              false, af, false, bf[mt], (short)0, zacc[mt], false, false);
      }
      // D-layout: VGPR v -> row n = nt*16 + hi8 + v, col m = mt*16 + laneM.
      // Per lane the 8 values are n-consecutive at fixed m: one 16B store into Z^T.
#pragma unroll
      for (int mt = 0; mt < 4; ++mt) {
        const int m = mt * 16 + laneM;
        v8bf zb = __builtin_convertvector(zacc[mt], v8bf);
        *(v8us*)(&sZT[m][nt * 16 + hi8]) = __builtin_bit_cast(v8us, zb);
      }
    }
    __syncthreads();

    // ---- Phase A2: mu partials (cheap VALU dot products from LDS)
#pragma unroll
    for (int pp = 0; pp < 2; ++pp) {
      const int p = tid + pp * 512;
      const int t = p >> 6, m = p & 63;
      const unsigned short* zp = &sZT[m][0];
      const unsigned short* wp = &sWb[t][0];
      float a = 0.0f;
      for (int n = 0; n < CHUNK; n += 8) {
        v8f zf = __builtin_convertvector(__builtin_bit_cast(v8bf, *(const v8us*)(zp + n)), v8f);
        v8f wf = __builtin_convertvector(__builtin_bit_cast(v8bf, *(const v8us*)(wp + n)), v8f);
#pragma unroll
        for (int j = 0; j < 8; ++j) a += wf[j] * zf[j];
      }
      sMu[p] += a;
    }

    // prefetch next chunk of Y while the heavy einsum runs
    {
      int cn = c + gridDim.x;
      if (cn < NCHUNK)
        __builtin_prefetch(Y + (size_t)cn * CHUNK * D_IN + (size_t)tid * 64, 0, 1);
    }

    // ---- Phase C: S[t] += Z^T * diag(w_t) * Z  (wave owns context t = wave)
    // A-frag(i) and the z-part of B-frag(i) are bit-identical per lane
    // (both read Z^T rows i*16+laneM over the same K run), so load once.
    {
      const unsigned short* wr = &sWb[wave][0];
#pragma unroll 2
      for (int kc = 0; kc < 8; ++kc) {
        const int n0 = kc * 32;
        v16bf zfr[4];
#pragma unroll
        for (int i = 0; i < 4; ++i)
          zfr[i] = ld_frag16(&sZT[i * 16 + laneM][0], n0, hi8);
        v16bf wfr = ld_frag16(wr, n0, hi8);   // broadcast along N lanes
        v16bf bfr[4];
#pragma unroll
        for (int kt = 0; kt < 4; ++kt) bfr[kt] = zfr[kt] * wfr;  // v_pk_mul_bf16
#pragma unroll
        for (int kt = 0; kt < 4; ++kt)
#pragma unroll
          for (int mt = 0; mt < 4; ++mt)
            acc[mt][kt] = __builtin_amdgcn_wmma_f32_16x16x32_bf16(
                false, zfr[mt], false, bfr[kt], (short)0, acc[mt][kt], false, false);
      }
    }
  }

  // ---- flush per-block partials (L2-resident f32 atomics)
  {
    float* Sb = S + (size_t)wave * (NMOD * NMOD);
#pragma unroll
    for (int mt = 0; mt < 4; ++mt)
#pragma unroll
      for (int kt = 0; kt < 4; ++kt)
#pragma unroll
        for (int v = 0; v < 8; ++v) {
          int m = mt * 16 + hi8 + v;
          int k = kt * 16 + laneM;
          atomicAdd(&Sb[m * NMOD + k], acc[mt][kt][v]);
        }
  }
  __syncthreads();
#pragma unroll
  for (int pp = 0; pp < 2; ++pp) {
    int p = tid + pp * 512;
    atomicAdd(&MU[p], sMu[p]);
  }
  if (tid < NCTX) atomicAdd(&WSUM[tid], sWsum[tid]);
}

__global__ __launch_bounds__(256) void mcspr_finalize_kernel(
    const float* __restrict__ S, const float* __restrict__ MU,
    const float* __restrict__ WSUM, const float* __restrict__ Cprior,
    const float* __restrict__ emaSigma, const unsigned char* __restrict__ emaInit,
    float* __restrict__ out) {
  __shared__ float sW[NCTX];
  __shared__ float sMu[NCTX * NMOD];
  __shared__ float sStd[NCTX * NMOD];
  __shared__ float sRed[256];
  const int tid = threadIdx.x;
  if (tid < NCTX) sW[tid] = WSUM[tid];
  __syncthreads();
  for (int p = tid; p < NCTX * NMOD; p += 256) sMu[p] = MU[p] / sW[p >> 6];
  __syncthreads();
  for (int p = tid; p < NCTX * NMOD; p += 256) {
    int t = p >> 6, m = p & 63;
    float sig = S[t * 4096 + m * 65] / sW[t] - sMu[p] * sMu[p];
    float emad = emaSigma[t * 4096 + m * 65];
    float sq = emaInit[t] ? (0.9f * emad + 0.1f * sig) : sig;   // BETA branch
    sStd[p] = sqrtf(sq + 1e-6f);
  }
  __syncthreads();
  float lacc = 0.0f;
  for (int idx = tid; idx < S_ELEMS; idx += 256) {
    int t = idx >> 12;
    int r = idx & 4095;
    int m = r >> 6, k = r & 63;
    float w = sW[t];
    if (w >= 30.0f) {                                           // K_MIN active mask
      float sig = S[idx] / w - sMu[t * 64 + m] * sMu[t * 64 + k];
      float den = sStd[t * 64 + m] * sStd[t * 64 + k] + 1e-6f;
      float ch = fminf(1.0f, fmaxf(-1.0f, sig / den));
      float d = ch - Cprior[idx];
      lacc += d * d / w;
    }
  }
  sRed[tid] = lacc;
  __syncthreads();
  for (int s = 128; s > 0; s >>= 1) {
    if (tid < s) sRed[tid] += sRed[tid + s];
    __syncthreads();
  }
  if (tid == 0) {
    int nact = 0;
    for (int t = 0; t < NCTX; ++t) nact += (sW[t] >= 30.0f) ? 1 : 0;
    out[0] = (nact > 0) ? (0.1f * 1.0f * sRed[0] / (float)nact) : 0.0f;
  }
}

extern "C" void kernel_launch(void* const* d_in, const int* in_sizes, int n_in,
                              void* d_out, int out_size, void* d_ws, size_t ws_size,
                              hipStream_t stream) {
  (void)in_sizes; (void)n_in; (void)out_size; (void)ws_size;
  const float* Y = (const float*)d_in[0];
  const float* CW = (const float*)d_in[1];
  const float* Mp = (const float*)d_in[2];
  const float* Cp = (const float*)d_in[3];
  const float* eS = (const float*)d_in[4];
  /* d_in[5] = ema_sigma_sq: unused by the reference math */
  const unsigned char* eI = (const unsigned char*)d_in[6];

  float* ws = (float*)d_ws;                 // needs WS_ELEMS*4 = 266 KB
  float* Sb = ws;
  float* MUb = ws + S_ELEMS;
  float* WSb = ws + S_ELEMS + MU_ELEMS;

  zero_ws_kernel<<<(WS_ELEMS + 255) / 256, 256, 0, stream>>>(ws);
  mcspr_accum_kernel<<<NBLOCKS, 512, 0, stream>>>(Y, CW, Mp, Sb, MUb, WSb);
  mcspr_finalize_kernel<<<1, 256, 0, stream>>>(Sb, MUb, WSb, Cp, eS, eI, (float*)d_out);
}
